// SCA_31610959298823
// MI455X (gfx1250) — matrix operand order repaired and checked by reading, not compiled, not run on hardware
//
#include <hip/hip_runtime.h>
#include <hip/hip_bf16.h>

typedef __attribute__((ext_vector_type(16))) __bf16 v16bf;
typedef __attribute__((ext_vector_type(8)))  float  v8f;
typedef __attribute__((ext_vector_type(2)))  float  v2f;
typedef __attribute__((ext_vector_type(8)))  unsigned int v8u;
typedef unsigned int u32;

#define BATCH 8
#define CH    256
#define SPAT  16384          // 128*128
#define CC    (CH*CH)        // 65536
#define KCH   64             // K-chunk staged in LDS per iteration

// ---------- helpers ----------
__device__ __forceinline__ float block_reduce_sum256(float v, float* s) {
    int t = threadIdx.x;
    s[t] = v; __syncthreads();
    #pragma unroll
    for (int o = 128; o > 0; o >>= 1) {
        if (t < o) s[t] += s[t + o];
        __syncthreads();
    }
    return s[0];
}

// async global->LDS copy of 16 bytes (ASYNCcnt-tracked, no VGPR staging)
__device__ __forceinline__ void async_b128(u32 lds_addr, const void* gptr) {
    unsigned long long ga = (unsigned long long)(uintptr_t)gptr;
    asm volatile("global_load_async_to_lds_b128 %0, %1, off"
                 :: "v"(lds_addr), "v"(ga) : "memory");
}

// ---------- K1: x -> bf16 copy + per-(b,c) mean ----------
__global__ void sca_prep(const float* __restrict__ x, __bf16* __restrict__ xb,
                         float* __restrict__ mean) {
    int row = blockIdx.x;                     // b*CH + c, 0..2047
    const float* xr = x  + (size_t)row * SPAT;
    __bf16*      xo = xb + (size_t)row * SPAT;
    float s = 0.f;
    for (int i = threadIdx.x; i < SPAT; i += 256) {
        float v = xr[i];
        s += v;
        xo[i] = (__bf16)v;
    }
    __shared__ float red[256];
    float tot = block_reduce_sum256(s, red);
    if (threadIdx.x == 0) mean[row] = tot * (1.0f / SPAT);
}

// ---------- K2: covariance via bf16 WMMA fed from async-LDS panels ----------
// Workgroup (8 waves) computes a 64x32 output tile; A(64xKCH) and B(32xKCH)
// bf16 panels double-buffered in LDS via global_load_async_to_lds_b128.
__global__ void sca_cov(const __bf16* __restrict__ xb, const float* __restrict__ mean,
                        float* __restrict__ cov) {
    int blk = blockIdx.x;                     // BATCH * 4 * 8
    int b   = blk >> 5;
    int it  = (blk >> 3) & 3;                 // i-tile (64 rows)
    int jt  = blk & 7;                        // j-tile (32 cols)
    int i0 = it * 64, j0 = jt * 32;

    __shared__ __bf16 lds[2][(64 + 32) * KCH];   // 2 x (8KB A + 4KB B)

    int t    = threadIdx.x;
    int lane = t & 31;
    int w    = t >> 5;                        // wave 0..7
    int wi   = w >> 1, wj = w & 1;            // 4x2 wave grid over the tile
    int m    = lane & 15, half = lane >> 4;

    const __bf16* base = xb + (size_t)b * CH * SPAT;

    // per-thread async-copy slices: A rows 0..63 (2 parts), B rows 0..31
    int ra0 = t >> 3;                         // 0..31
    int ka0 = (t & 7) * 8;                    // element offset within chunk
    const __bf16* ga0 = base + (size_t)(i0 + ra0)      * SPAT + ka0;
    const __bf16* ga1 = base + (size_t)(i0 + 32 + ra0) * SPAT + ka0;
    const __bf16* gb  = base + (size_t)(j0 + ra0)      * SPAT + ka0;

    u32 lA0[2], lA1[2], lB[2];
    #pragma unroll
    for (int buf = 0; buf < 2; ++buf) {
        lA0[buf] = (u32)(uintptr_t)&lds[buf][ra0 * KCH + ka0];
        lA1[buf] = (u32)(uintptr_t)&lds[buf][(32 + ra0) * KCH + ka0];
        lB [buf] = (u32)(uintptr_t)&lds[buf][64 * KCH + ra0 * KCH + ka0];
    }

    // pre-issue chunk 0
    async_b128(lA0[0], ga0);
    async_b128(lA1[0], ga1);
    async_b128(lB [0], gb );

    v8f acc = {};
    const int NCH = SPAT / KCH;               // 256 chunks
    for (int c = 0; c < NCH; ++c) {
        int buf = c & 1;
        if (c + 1 < NCH) {
            int nxt = buf ^ 1;
            size_t koff = (size_t)(c + 1) * KCH;
            async_b128(lA0[nxt], ga0 + koff);
            async_b128(lA1[nxt], ga1 + koff);
            async_b128(lB [nxt], gb  + koff);
            // 6 in flight per wave; current chunk done when <= 3 remain
            asm volatile("s_wait_asynccnt 0x3" ::: "memory");
        } else {
            asm volatile("s_wait_asynccnt 0x0" ::: "memory");
        }
        __syncthreads();

        const u32* ap = (const u32*)&lds[buf][(16 * wi + m) * KCH];
        const u32* bp = (const u32*)&lds[buf][64 * KCH + (16 * wj + m) * KCH];
        #pragma unroll
        for (int ks = 0; ks < 2; ++ks) {      // two K=32 WMMA steps per chunk
            union { v8u u; v16bf h; } ua, ub;
            int kd = ks * 16;                 // dword offset within the row
            // A 16x32 bf16 layout: lanes0-15 K=0..7 & 16..23, lanes16-31 K=8..15 & 24..31
            #pragma unroll
            for (int v = 0; v < 4; ++v) {
                ua.u[v]     = ap[kd + 4 * half + v];
                ua.u[v + 4] = ap[kd + 8 + 4 * half + v];
            }
            // B 32x16 bf16 layout: lanes0-15 K=0..7 & 16..23, lanes16-31 K=8..15 & 24..31
            #pragma unroll
            for (int v = 0; v < 4; ++v) {
                ub.u[v]     = bp[kd + 4 * half + v];
                ub.u[v + 4] = bp[kd + 8 + 4 * half + v];
            }
            acc = __builtin_amdgcn_wmma_f32_16x16x32_bf16(
                      false, ua.h, false, ub.h, (short)0, acc, false, false);
        }
        __syncthreads();
    }

    const float invM = 1.0f / SPAT;
    const float* mb = mean + b * CH;
    #pragma unroll
    for (int r = 0; r < 8; ++r) {
        int row = i0 + 16 * wi + r + 8 * half;   // C/D layout: vgpr r, lane-half
        int col = j0 + 16 * wj + m;
        float v = acc[r] * invM - mb[row] * mb[col];
        cov[((size_t)b * CH + row) * CH + col] = v;
    }
}

// ---------- K3: Frobenius norm per batch ----------
__global__ void sca_frob(const float* __restrict__ cov, float* __restrict__ normA) {
    int b = blockIdx.x;
    const float* cb = cov + (size_t)b * CC;
    float s = 0.f;
    for (int i = threadIdx.x; i < CC; i += 256) { float v = cb[i]; s += v * v; }
    __shared__ float red[256];
    float tot = block_reduce_sum256(s, red);
    if (threadIdx.x == 0) normA[b] = sqrtf(tot);
}

// ---------- K4: Y = cov / ||cov||_F ; Z = I ----------
__global__ void sca_init(const float* __restrict__ cov, const float* __restrict__ normA,
                         float* __restrict__ Y, float* __restrict__ Z) {
    int idx = blockIdx.x * 256 + threadIdx.x;   // < 8*65536
    int b  = idx >> 16;
    int rc = idx & 65535;
    float inv = 1.0f / normA[b];
    Y[idx] = cov[idx] * inv;
    Z[idx] = ((rc >> 8) == (rc & 255)) ? 1.0f : 0.0f;
}

// ---------- K5: batched 256x256 f32 GEMM via V_WMMA_F32_16X16X4_F32 ----------
// mode 0: D = A@B      mode 1: D = 1.5*I - 0.5*(A@B)
__global__ void sca_gemm256(const float* __restrict__ A, const float* __restrict__ B,
                            float* __restrict__ D, int mode) {
    int blk  = blockIdx.x;                    // 8 * 256
    int b    = blk >> 8;
    int tile = blk & 255;
    int ti = tile >> 4, tj = tile & 15;
    int lane = threadIdx.x & 31;
    int m    = lane & 15;
    int half = lane >> 4;
    int i0 = ti * 16, j0 = tj * 16;

    const float* Ab = A + (size_t)b * CC;
    const float* Bb = B + (size_t)b * CC;
    float*       Db = D + (size_t)b * CC;

    // A 16x4 f32 layout: lanes0-15 K=0,1 ; lanes16-31 K=2,3 (vgpr v -> K=2*half+v)
    const float* arow = Ab + (size_t)(i0 + m) * CH + 2 * half;
    // B 4x16 f32 layout (mirrored): lane n, K=2*half+v
    const float* bcol = Bb + (size_t)(2 * half) * CH + j0 + m;

    v8f acc = {};
    #pragma unroll 4
    for (int kb = 0; kb < CH; kb += 4) {
        v2f a, bv;
        a.x  = arow[kb];
        a.y  = arow[kb + 1];
        bv.x = bcol[(size_t)kb * CH];
        bv.y = bcol[(size_t)(kb + 1) * CH];
        acc = __builtin_amdgcn_wmma_f32_16x16x4_f32(
                  false, a, false, bv, (short)0, acc, false, false);
    }
    #pragma unroll
    for (int r = 0; r < 8; ++r) {
        int row = i0 + r + 8 * half;
        int col = j0 + m;
        float v = acc[r];
        if (mode == 1) v = ((row == col) ? 1.5f : 0.0f) - 0.5f * v;
        Db[(size_t)row * CH + col] = v;
    }
}

// ---------- K6: y[b][c] = mean_d Z[b][c][d] / sqrt(normA[b]) ----------
__global__ void sca_rowmean(const float* __restrict__ Z, const float* __restrict__ normA,
                            float* __restrict__ yv) {
    int row = blockIdx.x;                     // b*CH + c
    int b = row >> 8;
    float v = Z[(size_t)row * CH + threadIdx.x];
    __shared__ float red[256];
    float tot = block_reduce_sum256(v, red);
    if (threadIdx.x == 0)
        yv[row] = tot * (1.0f / CH) * rsqrtf(normA[b]);
}

// ---------- K7: squeeze-excite gate ----------
__global__ void sca_fc(const float* __restrict__ yv,
                       const float* __restrict__ w1, const float* __restrict__ b1,
                       const float* __restrict__ w2, const float* __restrict__ b2,
                       float* __restrict__ g) {
    int b = blockIdx.x;
    int t = threadIdx.x;
    __shared__ float ys[256];
    __shared__ float hs[32];
    ys[t] = yv[b * CH + t];
    __syncthreads();
    if (t < 32) {
        float acc = b1[t];
        #pragma unroll 8
        for (int c = 0; c < 256; ++c) acc += ys[c] * w1[t * 256 + c];
        hs[t] = fmaxf(acc, 0.0f);
    }
    __syncthreads();
    float acc = b2[t];
    #pragma unroll
    for (int j = 0; j < 32; ++j) acc += hs[j] * w2[t * 32 + j];
    g[b * CH + t] = 1.0f / (1.0f + __expf(-acc));
}

// ---------- K8: out = x * g (broadcast over HxW), float4 ----------
__global__ void sca_scale(const float4* __restrict__ x, const float* __restrict__ g,
                          float4* __restrict__ out) {
    size_t idx = (size_t)blockIdx.x * 256 + threadIdx.x;  // < 8,388,608
    int bc = (int)(idx >> 12);                // (idx*4)/16384
    float gg = g[bc];
    float4 v = x[idx];
    v.x *= gg; v.y *= gg; v.z *= gg; v.w *= gg;
    out[idx] = v;
}

extern "C" void kernel_launch(void* const* d_in, const int* in_sizes, int n_in,
                              void* d_out, int out_size, void* d_ws, size_t ws_size,
                              hipStream_t stream) {
    (void)in_sizes; (void)n_in; (void)out_size; (void)ws_size;
    const float* x  = (const float*)d_in[0];
    const float* w1 = (const float*)d_in[1];
    const float* b1 = (const float*)d_in[2];
    const float* w2 = (const float*)d_in[3];
    const float* b2 = (const float*)d_in[4];
    float* out = (float*)d_out;

    // workspace carve-out (256B aligned)
    char* ws = (char*)d_ws;
    size_t off = 0;
    auto carve = [&](size_t bytes) -> void* {
        off = (off + 255) & ~(size_t)255;
        void* p = ws + off;
        off += bytes;
        return p;
    };
    __bf16* xb    = (__bf16*)carve((size_t)BATCH * CH * SPAT * sizeof(__bf16)); // 64 MB
    float*  mean  = (float*) carve((size_t)BATCH * CH * sizeof(float));
    float*  cov   = (float*) carve((size_t)BATCH * CC * sizeof(float));
    float*  normA = (float*) carve((size_t)BATCH * sizeof(float));
    float*  Y0    = (float*) carve((size_t)BATCH * CC * sizeof(float));
    float*  Y1    = (float*) carve((size_t)BATCH * CC * sizeof(float));
    float*  Z0    = (float*) carve((size_t)BATCH * CC * sizeof(float));
    float*  Z1    = (float*) carve((size_t)BATCH * CC * sizeof(float));
    float*  T     = (float*) carve((size_t)BATCH * CC * sizeof(float));
    float*  yv    = (float*) carve((size_t)BATCH * CH * sizeof(float));
    float*  g     = (float*) carve((size_t)BATCH * CH * sizeof(float));

    // 1) bf16 copy + channel means
    sca_prep<<<BATCH * CH, 256, 0, stream>>>(x, xb, mean);
    // 2) covariance (bf16 WMMA from async-LDS double-buffered panels)
    sca_cov<<<BATCH * 32, 256, 0, stream>>>(xb, mean, cov);
    // 3) Frobenius norm per batch
    sca_frob<<<BATCH, 256, 0, stream>>>(cov, normA);
    // 4) Y = cov/||cov||, Z = I
    sca_init<<<(BATCH * CC) / 256, 256, 0, stream>>>(cov, normA, Y0, Z0);

    // 5) Newton-Schulz iterations (f32 WMMA)
    float* Yc = Y0; float* Yn = Y1;
    float* Zc = Z0; float* Zn = Z1;
    for (int it = 0; it < 5; ++it) {
        sca_gemm256<<<BATCH * 256, 32, 0, stream>>>(Zc, Yc, T, 1);  // T = 1.5I - 0.5 Z@Y
        sca_gemm256<<<BATCH * 256, 32, 0, stream>>>(Yc, T, Yn, 0);  // Y' = Y@T
        sca_gemm256<<<BATCH * 256, 32, 0, stream>>>(T, Zc, Zn, 0);  // Z' = T@Z
        float* tmp;
        tmp = Yc; Yc = Yn; Yn = tmp;
        tmp = Zc; Zc = Zn; Zn = tmp;
    }

    // 6) y = rowmean(Z)/sqrt(normA)
    sca_rowmean<<<BATCH * CH, 256, 0, stream>>>(Zc, normA, yv);
    // 7) gate
    sca_fc<<<BATCH, 256, 0, stream>>>(yv, w1, b1, w2, b2, g);
    // 8) rescale
    sca_scale<<<(BATCH * CH * SPAT) / (4 * 256), 256, 0, stream>>>(
        (const float4*)x, g, (float4*)out);
}
